// FFN_9062380995177
// MI455X (gfx1250) — matrix operand order repaired and checked by reading, not compiled, not run on hardware
//
#include <hip/hip_runtime.h>

typedef __attribute__((ext_vector_type(16))) __bf16 v16bf;
typedef __attribute__((ext_vector_type(8)))  float  v8f;

#define Bsz 32
#define Lsz 128
#define Vsz 512
#define Hsz 2048
#define LN_EPS 1e-3f

__device__ __forceinline__ v8f wmma_bf16(v16bf a, v16bf b, v8f c) {
  // D = A x B + C, fp32 accumulate -> v_wmma_f32_16x16x32_bf16
  return __builtin_amdgcn_wmma_f32_16x16x32_bf16(
      false, a, false, b, (short)0, c, false, false);
}

// Split-bf16: v ~= hi + lo with both parts bf16 (fp32-class product accuracy
// when accumulating hi*hi + hi*lo + lo*hi in fp32).
__device__ __forceinline__ void split_bf16(float v, __bf16& hi, __bf16& lo) {
  __bf16 h = (__bf16)v;
  hi = h;
  lo = (__bf16)(v - (float)h);
}

// ---------------------------------------------------------------------------
// Kernel 1: h[l,b,:] = relu(x[b,l,:] @ W1[l] + b1[l]), stored fp32 to d_ws
// grid = (H/128, L), block = 256 (8 waves, each owns 16 columns of H-tile)
// ---------------------------------------------------------------------------
__global__ void __launch_bounds__(256)
ffn_gemm1(const float* __restrict__ x, const float* __restrict__ W1,
          const float* __restrict__ b1, float* __restrict__ hws) {
  __shared__ __bf16 sXhi[Bsz * Vsz];        // 32 KB
  __shared__ __bf16 sXlo[Bsz * Vsz];        // 32 KB

  const int l    = blockIdx.y;
  const int hblk = blockIdx.x;              // 128 columns per block
  const int t    = threadIdx.x;

  // Stage x_l (32x512 fp32) -> split bf16 planes, coalesced along k
  for (int j = t; j < Bsz * Vsz; j += 256) {
    int m = j >> 9, k = j & 511;
    float v = x[(size_t)m * (Lsz * Vsz) + l * Vsz + k];
    split_bf16(v, sXhi[j], sXlo[j]);
  }
  __syncthreads();

  const int wave = t >> 5;
  const int lane = t & 31;
  const int lg   = lane >> 4;               // lane group (K-half select)
  const int ln   = lane & 15;
  const int ncol = hblk * 128 + wave * 16 + ln;   // this lane's N column

  const float* W1l = W1 + (size_t)l * Vsz * Hsz;

  v8f acc0 = {};                            // rows 0..15
  v8f acc1 = {};                            // rows 16..31

  for (int kb = 0; kb < Vsz; kb += 32) {
    // A fragments (16-bit A 16x32 layout): group0 K={0..7,16..23}, group1 +8
    v16bf ahi0, alo0, ahi1, alo1;
#pragma unroll
    for (int j = 0; j < 16; ++j) {
      int k = kb + j + lg * 8 + ((j & 8) ? 8 : 0);
      ahi0[j] = sXhi[ln * Vsz + k];
      alo0[j] = sXlo[ln * Vsz + k];
      ahi1[j] = sXhi[(ln + 16) * Vsz + k];
      alo1[j] = sXlo[(ln + 16) * Vsz + k];
    }
    // B fragment (32x16): lanes 0-15 K={0..7,16..23}, lanes 16-31 +8
    v16bf bhi, blo;
#pragma unroll
    for (int j = 0; j < 16; ++j) {
      int k = kb + j + lg * 8 + ((j & 8) ? 8 : 0);
      float w = W1l[(size_t)k * Hsz + ncol];    // coalesced across lanes
      __bf16 h, lo; split_bf16(w, h, lo);
      bhi[j] = h; blo[j] = lo;
    }
    acc0 = wmma_bf16(alo0, bhi, acc0);
    acc0 = wmma_bf16(ahi0, blo, acc0);
    acc0 = wmma_bf16(ahi0, bhi, acc0);
    acc1 = wmma_bf16(alo1, bhi, acc1);
    acc1 = wmma_bf16(ahi1, blo, acc1);
    acc1 = wmma_bf16(ahi1, bhi, acc1);
  }

  // Epilogue: D layout m = lg*8 + r (+16 for acc1), n = ln
  const float bias = b1[l * Hsz + ncol];
#pragma unroll
  for (int r = 0; r < 8; ++r) {
    int m0 = lg * 8 + r;
    float v0 = acc0[r] + bias; v0 = v0 > 0.f ? v0 : 0.f;
    float v1 = acc1[r] + bias; v1 = v1 > 0.f ? v1 : 0.f;
    hws[((size_t)l * Bsz + m0) * Hsz + ncol]      = v0;
    hws[((size_t)l * Bsz + m0 + 16) * Hsz + ncol] = v1;
  }
}

// ---------------------------------------------------------------------------
// Kernel 2: out = LayerNorm(h[l] @ W2[l] + b2[l] + x[:,l,:]) per row
// grid = L, block = 256. Each workgroup owns the full 32x512 output tile.
// ---------------------------------------------------------------------------
__global__ void __launch_bounds__(256)
ffn_gemm2_ln(const float* __restrict__ hws, const float* __restrict__ W2,
             const float* __restrict__ b2, const float* __restrict__ x,
             const float* __restrict__ gamma, const float* __restrict__ beta,
             float* __restrict__ out) {
  __shared__ float sRes[Bsz * Vsz];         // 64 KB fp32 residual buffer
  __shared__ float sPartS[Bsz][8];
  __shared__ float sPartQ[Bsz][8];
  __shared__ float sMean[Bsz];
  __shared__ float sRstd[Bsz];
  // Alias the residual buffer as the split h-chunk planes during the K loop:
  // [0, 32KB) = hi plane, [32KB, 64KB) = lo plane (32x512 bf16 each).
  __bf16* sHhi = (__bf16*)sRes;
  __bf16* sHlo = ((__bf16*)sRes) + Bsz * 512;

  const int l    = blockIdx.x;
  const int t    = threadIdx.x;
  const int wave = t >> 5;
  const int lane = t & 31;
  const int lg   = lane >> 4;
  const int ln   = lane & 15;

  v8f acc[4][2];
#pragma unroll
  for (int p = 0; p < 4; ++p) { acc[p][0] = (v8f){}; acc[p][1] = (v8f){}; }

  const float* W2l = W2 + (size_t)l * Hsz * Vsz;
  const float* hl  = hws + (size_t)l * Bsz * Hsz;

  for (int kc = 0; kc < 4; ++kc) {          // K chunks of 512
    for (int j = t; j < Bsz * 512; j += 256) {
      int m = j >> 9, kk = j & 511;
      float v = hl[(size_t)m * Hsz + kc * 512 + kk];
      split_bf16(v, sHhi[j], sHlo[j]);
    }
    __syncthreads();

    for (int ks = 0; ks < 512; ks += 32) {
      v16bf ahi0, alo0, ahi1, alo1;
#pragma unroll
      for (int j = 0; j < 16; ++j) {
        int k = ks + j + lg * 8 + ((j & 8) ? 8 : 0);
        ahi0[j] = sHhi[ln * 512 + k];
        alo0[j] = sHlo[ln * 512 + k];
        ahi1[j] = sHhi[(ln + 16) * 512 + k];
        alo1[j] = sHlo[(ln + 16) * 512 + k];
      }
#pragma unroll
      for (int p = 0; p < 4; ++p) {
        int n = p * 128 + wave * 16 + ln;
        v16bf bhi, blo;
#pragma unroll
        for (int j = 0; j < 16; ++j) {
          int k = kc * 512 + ks + j + lg * 8 + ((j & 8) ? 8 : 0);
          float w = W2l[(size_t)k * Vsz + n];
          __bf16 h, lo; split_bf16(w, h, lo);
          bhi[j] = h; blo[j] = lo;
        }
        acc[p][0] = wmma_bf16(alo0, bhi, acc[p][0]);
        acc[p][0] = wmma_bf16(ahi0, blo, acc[p][0]);
        acc[p][0] = wmma_bf16(ahi0, bhi, acc[p][0]);
        acc[p][1] = wmma_bf16(alo1, bhi, acc[p][1]);
        acc[p][1] = wmma_bf16(ahi1, blo, acc[p][1]);
        acc[p][1] = wmma_bf16(ahi1, bhi, acc[p][1]);
      }
    }
    __syncthreads();                        // protect sH planes before next stage
  }

  // Residual (+bias +x) into fp32 LDS, D-matrix layout
#pragma unroll
  for (int p = 0; p < 4; ++p) {
    int n = p * 128 + wave * 16 + ln;
    float bb = b2[l * Vsz + n];
#pragma unroll
    for (int r = 0; r < 8; ++r) {
      int m0 = lg * 8 + r;
      float r0 = acc[p][0][r] + bb + x[(size_t)m0 * (Lsz * Vsz) + l * Vsz + n];
      float r1 = acc[p][1][r] + bb + x[(size_t)(m0 + 16) * (Lsz * Vsz) + l * Vsz + n];
      sRes[m0 * Vsz + n]        = r0;
      sRes[(m0 + 16) * Vsz + n] = r1;
    }
  }
  __syncthreads();

  // LayerNorm over V=512 per row: 8 threads per row, two-stage reduction
  {
    int row = t >> 3, seg = t & 7;
    float s = 0.f, q = 0.f;
    for (int c = seg * 64; c < seg * 64 + 64; ++c) {
      float v = sRes[row * Vsz + c];
      s += v; q += v * v;
    }
    sPartS[row][seg] = s;
    sPartQ[row][seg] = q;
  }
  __syncthreads();
  if (t < Bsz) {
    float s = 0.f, q = 0.f;
#pragma unroll
    for (int i = 0; i < 8; ++i) { s += sPartS[t][i]; q += sPartQ[t][i]; }
    float mean = s * (1.0f / Vsz);
    float var  = q * (1.0f / Vsz) - mean * mean;
    sMean[t] = mean;
    sRstd[t] = rsqrtf(var + LN_EPS);
  }
  __syncthreads();
  {
    int row = t >> 3, seg = t & 7;
    float mean = sMean[row], rstd = sRstd[row];
    for (int c = seg * 64; c < seg * 64 + 64; ++c) {
      float v = (sRes[row * Vsz + c] - mean) * rstd * gamma[c] + beta[c];
      out[(size_t)row * (Lsz * Vsz) + l * Vsz + c] = v;
    }
  }
}

// ---------------------------------------------------------------------------
extern "C" void kernel_launch(void* const* d_in, const int* in_sizes, int n_in,
                              void* d_out, int out_size, void* d_ws, size_t ws_size,
                              hipStream_t stream) {
  const float* x     = (const float*)d_in[0];
  const float* W1    = (const float*)d_in[1];
  const float* b1    = (const float*)d_in[2];
  const float* W2    = (const float*)d_in[3];
  const float* b2    = (const float*)d_in[4];
  const float* gamma = (const float*)d_in[5];
  const float* beta  = (const float*)d_in[6];
  float* out = (float*)d_out;
  float* hws = (float*)d_ws;                // L*B*H fp32 = 32 MB scratch

  ffn_gemm1<<<dim3(Hsz / 128, Lsz), 256, 0, stream>>>(x, W1, b1, hws);
  ffn_gemm2_ln<<<dim3(Lsz), 256, 0, stream>>>(hws, W2, b2, x, gamma, beta, out);
}
